// SelfAttnLayerV2_78623671320669
// MI455X (gfx1250) — compile-verified, matrix operand order checked by reading, not run on hardware
//
#include <hip/hip_runtime.h>
#include <hip/hip_bf16.h>

// ---------------------------------------------------------------------------
// SelfAttn fused kernel for MI455X (gfx1250, wave32, WMMA)
//   B=16, C=256, H=W=48 -> N=2304, DQK=32
//   q = Wq x + bq ; k = Wk x + bk ; v = Wv x + bv
//   A = softmax(q k^T) ; out = gamma * (V A^T) + x
// Flash-attention fusion with v_wmma_f32_16x16x32_bf16. S is computed
// TRANSPOSED (S^T = K Q^T) so that the D-register layout of S^T coincides
// with the A-fragment layout needed by the O-update WMMA: softmax runs
// in-lane and exp(S) packs directly into the next WMMA's A operand.
// ---------------------------------------------------------------------------

typedef __attribute__((ext_vector_type(16))) __bf16 v16bf;
typedef __attribute__((ext_vector_type(8)))  float  v8f;

static constexpr int Bn = 16;
static constexpr int Cn = 256;
static constexpr int Nn = 48 * 48;   // 2304
static constexpr int Dq = 32;        // C/8

union AFrag {
    uint4          u[2];
    unsigned short s[16];
    v16bf          v;
};

static __device__ __forceinline__ unsigned short f32_to_bf16(float f) {
    unsigned u = __float_as_uint(f);
    unsigned r = u + 0x7FFFu + ((u >> 16) & 1u);   // round-to-nearest-even
    return (unsigned short)(r >> 16);
}
static __device__ __forceinline__ unsigned pack_bf16x2(float a, float b) {
    return (unsigned)f32_to_bf16(a) | ((unsigned)f32_to_bf16(b) << 16);
}

// ---------------------------------------------------------------------------
// Projection: Q and K. One thread per (b,n); weights transposed into LDS so
// the c-loop does broadcast LDS reads; x reads are coalesced along n.
// Qb/Kb layout: [B][N][32] bf16 row-major (row = one query/key vector).
// ---------------------------------------------------------------------------
__global__ __launch_bounds__(256) void proj_qk_kernel(
    const float* __restrict__ x,
    const float* __restrict__ wq, const float* __restrict__ bq,
    const float* __restrict__ wk, const float* __restrict__ bk,
    unsigned short* __restrict__ Qb, unsigned short* __restrict__ Kb)
{
    __shared__ float wqs[Dq * Cn];   // [c][d]
    __shared__ float wks[Dq * Cn];
    const int tid = threadIdx.x;
    for (int idx = tid; idx < Dq * Cn; idx += 256) {
        int d = idx / Cn, c = idx % Cn;
        wqs[c * Dq + d] = wq[idx];
        wks[c * Dq + d] = wk[idx];
    }
    __syncthreads();

    const int gn = blockIdx.x * 256 + tid;   // flat (b,n), exact cover
    const int b  = gn / Nn;
    const int n  = gn % Nn;

    float aq[Dq], ak[Dq];
    #pragma unroll
    for (int d = 0; d < Dq; ++d) { aq[d] = bq[d]; ak[d] = bk[d]; }

    const float* xp = x + (size_t)b * Cn * Nn + n;
    for (int c = 0; c < Cn; ++c) {
        const float xv = xp[(size_t)c * Nn];
        const float* wqr = &wqs[c * Dq];
        const float* wkr = &wks[c * Dq];
        #pragma unroll
        for (int d = 0; d < Dq; ++d) {
            aq[d] = fmaf(wqr[d], xv, aq[d]);
            ak[d] = fmaf(wkr[d], xv, ak[d]);
        }
    }
    unsigned* qdst = (unsigned*)(Qb + (size_t)gn * Dq);
    unsigned* kdst = (unsigned*)(Kb + (size_t)gn * Dq);
    #pragma unroll
    for (int d = 0; d < Dq / 2; ++d) {
        qdst[d] = pack_bf16x2(aq[2 * d], aq[2 * d + 1]);
        kdst[d] = pack_bf16x2(ak[2 * d], ak[2 * d + 1]);
    }
}

// ---------------------------------------------------------------------------
// Projection: V. Block = (b, 64-wide n tile); x tile [256c][64n] in LDS
// (64 KB), thread tid owns output channel c=tid, accumulates 64 n values.
// Vb layout: [B][C][N] bf16 (j contiguous -> contiguous B-fragment loads).
// ---------------------------------------------------------------------------
__global__ __launch_bounds__(256) void proj_v_kernel(
    const float* __restrict__ x,
    const float* __restrict__ wv, const float* __restrict__ bv,
    unsigned short* __restrict__ Vb)
{
    __shared__ float xs[Cn * 64];
    const int b   = blockIdx.y;
    const int n0  = blockIdx.x * 64;
    const int tid = threadIdx.x;

    // cooperative coalesced fill: 4 threads per channel row, 4 passes
    {
        const int r  = tid >> 2;
        const int cq = (tid & 3) * 16;
        for (int pass = 0; pass < 4; ++pass) {
            const int c = pass * 64 + r;
            const float4* src = (const float4*)(x + ((size_t)b * Cn + c) * Nn + n0 + cq);
            float4* dst = (float4*)(&xs[c * 64 + cq]);
            #pragma unroll
            for (int j = 0; j < 4; ++j) dst[j] = src[j];
        }
    }
    __syncthreads();

    const int co = tid;
    float acc[64];
    {
        const float bias = bv[co];
        #pragma unroll
        for (int n = 0; n < 64; ++n) acc[n] = bias;
    }
    const float* wr = wv + (size_t)co * Cn;
    for (int cc = 0; cc < Cn; ++cc) {
        const float w = wr[cc];
        const float4* xr = (const float4*)(&xs[cc * 64]);  // broadcast reads
        #pragma unroll
        for (int n4 = 0; n4 < 16; ++n4) {
            float4 xv = xr[n4];
            acc[n4 * 4 + 0] = fmaf(w, xv.x, acc[n4 * 4 + 0]);
            acc[n4 * 4 + 1] = fmaf(w, xv.y, acc[n4 * 4 + 1]);
            acc[n4 * 4 + 2] = fmaf(w, xv.z, acc[n4 * 4 + 2]);
            acc[n4 * 4 + 3] = fmaf(w, xv.w, acc[n4 * 4 + 3]);
        }
    }
    unsigned* dst = (unsigned*)(Vb + ((size_t)b * Cn + co) * Nn + n0);
    #pragma unroll
    for (int n = 0; n < 32; ++n)
        dst[n] = pack_bf16x2(acc[2 * n], acc[2 * n + 1]);
}

// ---------------------------------------------------------------------------
// Flash attention. Block = 128 threads (4 waves); wave owns 16 query rows.
//
// Fragment layouts (ISA 7.12.2), lane L, half h=L/16, lm=L%16, kb=8h:
//   16-bit A 16x32 : row lm,  K in {kb..kb+7} u {16+kb..16+kb+7}
//   16-bit B 32x16 : col lm,  K same pattern
//   f32   D 16x16  : (M,N) = (v + 8h, lm), v = VGPR index
//
// Per 32-col block jb:
//   S^T tiles  = wmma(A=K rows, B=Q^T)  -> lane holds 16 S values of query
//                row i=lm (j = jb + {kb..kb+7} u {16+kb..16+kb+7})
//   softmax    : in-lane reduce + tiny per-wave LDS combine of the 2 halves
//   P fragment : exp() packs bf16 DIRECTLY into A layout (no relayout!)
//   O update   : 16x wmma, O[16,256] f32 in VGPRs, rescale only when the
//                running max changed (wave-uniform __any guard)
// ---------------------------------------------------------------------------
__global__ __launch_bounds__(128) void flash_attn_kernel(
    const unsigned short* __restrict__ Qb,
    const unsigned short* __restrict__ Kb,
    const unsigned short* __restrict__ Vb,
    const float* __restrict__ x,
    const float* __restrict__ gamma,
    float* __restrict__ out)
{
    __shared__ alignas(16) float stat[4][2][16];  // per-wave half-combine
    __shared__ alignas(16) float rowb[4][16];     // per-wave row broadcast

    const int b    = blockIdx.y;
    const int i0wg = blockIdx.x * 64;
    const int tid  = threadIdx.x;
    const int wave = tid >> 5;
    const int lane = tid & 31;
    const int half = lane >> 4;
    const int lm   = lane & 15;
    const int kb   = half * 8;
    const int i0   = i0wg + wave * 16;

    // Q^T fragment (B-matrix): column i = lm, K(d) pattern kb
    AFrag bQ;
    {
        const unsigned short* qrow = Qb + ((size_t)b * Nn + i0 + lm) * Dq;
        bQ.u[0] = *(const uint4*)(qrow + kb);
        bQ.u[1] = *(const uint4*)(qrow + 16 + kb);
    }

    float m = -3.0e38f;   // running max for query row i = i0 + lm
    float l = 0.0f;       // running sum for query row i = i0 + lm
    v8f o[16];
    #pragma unroll
    for (int t = 0; t < 16; ++t)
        #pragma unroll
        for (int v = 0; v < 8; ++v) o[t][v] = 0.0f;

    const v8f zero8 = {0.f, 0.f, 0.f, 0.f, 0.f, 0.f, 0.f, 0.f};

    for (int jb = 0; jb < Nn; jb += 32) {
        // --- K A-fragments: rows jb..jb+31 ---
        const unsigned short* krow = Kb + ((size_t)b * Nn + jb + lm) * Dq;
        AFrag aK0, aK1;
        aK0.u[0] = *(const uint4*)(krow + kb);
        aK0.u[1] = *(const uint4*)(krow + 16 + kb);
        aK1.u[0] = *(const uint4*)(krow + 16 * Dq + kb);
        aK1.u[1] = *(const uint4*)(krow + 16 * Dq + 16 + kb);
        if (jb + 32 < Nn) __builtin_prefetch(krow + 32 * Dq, 0, 3);

        // S^T tiles: lane holds S[i=lm][jb + kb+v] (S0) and [jb+16+kb+v] (S1)
        v8f S0 = __builtin_amdgcn_wmma_f32_16x16x32_bf16(
            false, aK0.v, false, bQ.v, (short)0, zero8, false, false);
        v8f S1 = __builtin_amdgcn_wmma_f32_16x16x32_bf16(
            false, aK1.v, false, bQ.v, (short)0, zero8, false, false);

        // --- in-lane max over this lane's 16 values, combine halves in LDS --
        float pm = fmaxf(S0[0], S1[0]);
        #pragma unroll
        for (int v = 1; v < 8; ++v) pm = fmaxf(pm, fmaxf(S0[v], S1[v]));
        stat[wave][half][lm] = pm;
        const float rm = fmaxf(stat[wave][0][lm], stat[wave][1][lm]);
        const float nm = fmaxf(m, rm);
        const bool  changed = nm > m;

        // --- exp + pack straight into the A fragment of the O update --------
        AFrag aP;
        float ps = 0.0f;
        #pragma unroll
        for (int v = 0; v < 8; ++v) {
            float p0 = __expf(S0[v] - nm);
            float p1 = __expf(S1[v] - nm);
            ps += p0 + p1;
            aP.s[v]     = f32_to_bf16(p0);
            aP.s[8 + v] = f32_to_bf16(p1);
        }
        stat[wave][half][lm] = ps;
        const float rs = stat[wave][0][lm] + stat[wave][1][lm];
        const float sc = __expf(m - nm);
        l = l * sc + rs;
        m = nm;

        // --- rescale O only when the running max moved (wave-uniform) -------
        if (__any(changed)) {
            if (half == 0) rowb[wave][lm] = sc;       // row r lives in lane r
            const float4 s0 = *(const float4*)(&rowb[wave][8 * half]);
            const float4 s1 = *(const float4*)(&rowb[wave][8 * half + 4]);
            const float sc8[8] = {s0.x, s0.y, s0.z, s0.w, s1.x, s1.y, s1.z, s1.w};
            #pragma unroll
            for (int t = 0; t < 16; ++t)
                #pragma unroll
                for (int v = 0; v < 8; ++v) o[t][v] *= sc8[v];
        }

        // --- O += P * V' : 16 channel tiles (V is L2-resident, 18.9 MB) -----
        #pragma unroll
        for (int t = 0; t < 16; ++t) {
            const unsigned short* vrow =
                Vb + ((size_t)b * Cn + t * 16 + lm) * Nn + jb;
            AFrag bV;
            bV.u[0] = *(const uint4*)(vrow + kb);
            bV.u[1] = *(const uint4*)(vrow + 16 + kb);
            o[t] = __builtin_amdgcn_wmma_f32_16x16x32_bf16(
                false, aP.v, false, bV.v, (short)0, o[t], false, false);
        }
    }

    // --- epilogue: out[b][c][i] = gamma * O[i,c]/l[i] + x[b][c][i] ----------
    if (half == 0) rowb[wave][lm] = l;
    const float4 l0 = *(const float4*)(&rowb[wave][8 * half]);
    const float4 l1 = *(const float4*)(&rowb[wave][8 * half + 4]);
    float rl[8] = {1.0f / l0.x, 1.0f / l0.y, 1.0f / l0.z, 1.0f / l0.w,
                   1.0f / l1.x, 1.0f / l1.y, 1.0f / l1.z, 1.0f / l1.w};
    const float g = gamma[0];

    #pragma unroll
    for (int t = 0; t < 16; ++t) {
        const int c = t * 16 + lm;
        const size_t cbase = ((size_t)b * Cn + c) * Nn;
        #pragma unroll
        for (int v = 0; v < 8; ++v) {
            const int i = i0 + v + 8 * half;
            const size_t gi = cbase + i;
            out[gi] = g * (o[t][v] * rl[v]) + x[gi];
        }
    }
}

// ---------------------------------------------------------------------------
// Launch
// ---------------------------------------------------------------------------
extern "C" void kernel_launch(void* const* d_in, const int* in_sizes, int n_in,
                              void* d_out, int out_size, void* d_ws, size_t ws_size,
                              hipStream_t stream) {
    (void)in_sizes; (void)n_in; (void)out_size; (void)ws_size;
    const float* x     = (const float*)d_in[0];
    const float* wq    = (const float*)d_in[1];
    const float* bq    = (const float*)d_in[2];
    const float* wk    = (const float*)d_in[3];
    const float* bk    = (const float*)d_in[4];
    const float* wv    = (const float*)d_in[5];
    const float* bv    = (const float*)d_in[6];
    const float* gamma = (const float*)d_in[7];
    float* out = (float*)d_out;

    // workspace: Qb | Kb | Vb (bf16)  -> ~23.6 MB
    unsigned short* Qb = (unsigned short*)d_ws;
    unsigned short* Kb = Qb + (size_t)Bn * Nn * Dq;
    unsigned short* Vb = Kb + (size_t)Bn * Nn * Dq;

    proj_qk_kernel<<<dim3((Bn * Nn) / 256), dim3(256), 0, stream>>>(
        x, wq, bq, wk, bk, Qb, Kb);
    proj_v_kernel<<<dim3(Nn / 64, Bn), dim3(256), 0, stream>>>(
        x, wv, bv, Vb);
    flash_attn_kernel<<<dim3(Nn / 64, Bn), dim3(128), 0, stream>>>(
        Qb, Kb, Vb, x, gamma, out);
}
